// LSTMAtt_Classifier_54142357733604
// MI455X (gfx1250) — matrix operand order repaired and checked by reading, not compiled, not run on hardware
//
#include <hip/hip_runtime.h>
#include <hip/hip_bf16.h>

typedef __attribute__((ext_vector_type(16))) _Float16 v16h;
typedef __attribute__((ext_vector_type(8)))  float    v8f;

#define Bn 8
#define Ln 512
#define Dn 32
#define Nn 64
#define Hn 64

// workspace byte offsets (all 256B aligned)
#define OFF_ASW     0u            // A swizzled fragments: B*32*16*512 f16 = 4194304 B
#define OFF_BSW     4194304u      // B swizzled fragments: B*16*2*512 f16 = 262144 B
#define OFF_WX      4456448u      // B*L*N f32 = 1048576 B
#define OFF_WXHAT   5505024u      // B*L*N f32 = 1048576 B
#define OFF_XATT    6553600u      // B*L*D f32 = 524288 B

__device__ __forceinline__ float sigmoidf_(float x) {
    return 1.0f / (1.0f + __expf(-x));
}

// ---------------- K0: build swizzled B fragments from f32 text ----------------
// Layout: Bsw[((b*16 + kc)*2 + wv)*512 + lane*16 + e]
//   lane = h*16 + (n - 16*wv),  element e -> K = kc*32 + 8*h + (e&7) + 16*(e>>3), column n.
__global__ __launch_bounds__(256) void swizzle_b(const float* __restrict__ text,
                                                 _Float16* __restrict__ Bsw) {
    int idx = blockIdx.x * 256 + threadIdx.x;          // B*16*2*32*16 = 131072
    if (idx >= Bn * 16 * 2 * 32 * 16) return;
    int e    = idx & 15;
    int lane = (idx >> 4) & 31;
    int wv   = (idx >> 9) & 1;
    int kc   = (idx >> 10) & 15;
    int b    = idx >> 14;
    int n = wv * 16 + (lane & 15);
    int h = lane >> 4;
    int k = kc * 32 + 8 * h + (e & 7) + ((e & 8) << 1);
    Bsw[idx] = (_Float16)text[((size_t)b * Ln + k) * Dn + n];
}

// ---------------- K1: Wx / Wxhat projections ----------------
__global__ __launch_bounds__(256) void proj(const float* __restrict__ text,
                                            const float* __restrict__ Wx_w,
                                            const float* __restrict__ Wx_b,
                                            const float* __restrict__ Wxh_w,
                                            const float* __restrict__ Wxh_b,
                                            float* __restrict__ Wx,
                                            float* __restrict__ Wxh) {
    int gid = blockIdx.x * 256 + threadIdx.x;
    if (gid >= Bn * Ln * Nn) return;
    int n  = gid & (Nn - 1);
    int bl = gid >> 6;                       // flat (b,l), 0..4095
    const float* x = text + (size_t)bl * Dn;
    float a1 = 0.f, a2 = 0.f;
#pragma unroll
    for (int d = 0; d < Dn; ++d) {
        float xv = x[d];
        a1 += xv * Wx_w [n * Dn + d];
        a2 += xv * Wxh_w[n * Dn + d];
    }
    Wx [gid] = a1 + Wx_b [n];
    Wxh[gid] = a2 + Wxh_b[n];
}

// ---------------- K2: attention scores, stored directly as WMMA A-fragments ----------------
// Asw[(((b*32 + iT)*16 + kc)*32 + h*16 + m)*16 + e]
//   element (i,j): m=i&15, iT=i>>4; kw=j&31, kc=j>>5; h=(kw>>3)&1; e=(kw&7)|((kw&16)>>1)
__global__ __launch_bounds__(256) void attn_scores(const float* __restrict__ Wx,
                                                   const float* __restrict__ Wxh,
                                                   const float* __restrict__ att_w,
                                                   const float* __restrict__ att_b,
                                                   _Float16* __restrict__ Asw) {
    __shared__ float wxS[16][Nn];
    __shared__ float whS[16][Nn];
    __shared__ float awS[Nn];
    const int b  = blockIdx.z;
    const int iT = blockIdx.y;               // i0 = iT*16
    const int j0 = blockIdx.x * 16;
    const int tid = threadIdx.x;
#pragma unroll
    for (int r = 0; r < 4; ++r) {
        int idx = tid + 256 * r;
        int row = idx >> 6, col = idx & 63;
        wxS[row][col] = Wx [((size_t)b * Ln + iT * 16 + row) * Nn + col];
        whS[row][col] = Wxh[((size_t)b * Ln + j0 + row) * Nn + col];
    }
    if (tid < Nn) awS[tid] = att_w[tid];
    __syncthreads();

    const int il = tid >> 4, jl = tid & 15;
    float acc = att_b[0];
#pragma unroll 4
    for (int n = 0; n < Nn; ++n)
        acc += awS[n] * tanhf(whS[jl][n] + wxS[il][n]);
    float Aval = sigmoidf_(acc);

    const int j  = j0 + jl;
    const int kc = j >> 5;
    const int kw = j & 31;
    const int h  = (kw >> 3) & 1;
    const int e  = (kw & 7) | ((kw & 16) >> 1);
    Asw[((((size_t)b * 32 + iT) * 16 + kc) * 32 + h * 16 + il) * 16 + e] = (_Float16)Aval;
}

// ---------------- K3: Xatt = A @ text via v_wmma_f32_16x16x32_f16 ----------------
// block = 64 threads = 2 waves; wave wv handles n-tile n0 = 16*wv. Operands pre-swizzled:
// per k-chunk each lane loads 32 contiguous bytes of A and of B (b128 pairs).
__global__ __launch_bounds__(64) void xatt_wmma(const _Float16* __restrict__ Asw,
                                                const _Float16* __restrict__ Bsw,
                                                float* __restrict__ Xatt) {
    const int b    = blockIdx.y;
    const int iT   = blockIdx.x;
    const int lane = threadIdx.x & 31;
    const int wv   = threadIdx.x >> 5;
    const int h    = lane >> 4;
    const int n    = wv * 16 + (lane & 15);

    const _Float16* aBase = Asw + (((size_t)b * 32 + iT) * 16) * 512 + (size_t)lane * 16;
    const _Float16* bBase = Bsw + ((size_t)b * 16 * 2 + wv) * 512 + (size_t)lane * 16;

    v8f c = {};
#pragma unroll 4
    for (int kc = 0; kc < Ln / 32; ++kc) {
        v16h a = *(const v16h*)(aBase + (size_t)kc * 512);
        v16h bm = *(const v16h*)(bBase + (size_t)kc * 1024);
        c = __builtin_amdgcn_wmma_f32_16x16x32_f16(
                /*neg_a=*/false, a, /*neg_b=*/false, bm,
                /*c_mod=*/(short)0, c, /*reuse_a=*/false, /*reuse_b=*/false);
    }
    // D layout: VGPR r -> row m = r + 8h, column n
#pragma unroll
    for (int r = 0; r < 8; ++r)
        Xatt[((size_t)b * Ln + (iT * 16 + r + 8 * h)) * Dn + n] = c[r];
}

// ---------------- K4: LSTM scan, one block per batch, + dense head ----------------
__global__ __launch_bounds__(256) void lstm_scan(const float* __restrict__ Xatt,
                                                 const float* __restrict__ W_ih,
                                                 const float* __restrict__ W_hh,
                                                 const float* __restrict__ b_ih,
                                                 const float* __restrict__ b_hh,
                                                 const float* __restrict__ dense_w,
                                                 const float* __restrict__ dense_b,
                                                 float* __restrict__ out) {
    __shared__ float hS[Hn];
    __shared__ float cS[Hn];
    __shared__ float xS[Dn];
    __shared__ float gS[4 * Hn];
    const int b   = blockIdx.x;
    const int tid = threadIdx.x;          // owns gate column g = tid

    if (tid < Hn) { hS[tid] = 0.f; cS[tid] = 0.f; }
    if (tid < Dn) xS[tid] = Xatt[(size_t)b * Ln * Dn + tid];     // x(0)

    // this gate column's weight rows in registers (96 f32)
    float wih[Dn];
    float whh[Hn];
#pragma unroll
    for (int d = 0; d < Dn; ++d) wih[d] = W_ih[tid * Dn + d];
#pragma unroll
    for (int k = 0; k < Hn; ++k) whh[k] = W_hh[tid * Hn + k];
    const float bsum = b_ih[tid] + b_hh[tid];

    for (int t = 0; t < Ln; ++t) {
        __syncthreads();                  // h(t-1) updated, xS = x(t)
        float acc = bsum;
#pragma unroll
        for (int d = 0; d < Dn; ++d) acc += wih[d] * xS[d];
#pragma unroll
        for (int k = 0; k < Hn; ++k) acc += whh[k] * hS[k];
        gS[tid] = acc;
        __syncthreads();                  // gates ready; xS/hS reads done
        if (tid < Dn && t + 1 < Ln)
            xS[tid] = Xatt[((size_t)b * Ln + t + 1) * Dn + tid]; // prefetch x(t+1)
        if (tid < Hn) {
            float si = sigmoidf_(gS[tid]);
            float sf = sigmoidf_(gS[tid + Hn]);
            float gg = tanhf   (gS[tid + 2 * Hn]);
            float so = sigmoidf_(gS[tid + 3 * Hn]);
            float cv = sf * cS[tid] + si * gg;
            cS[tid] = cv;
            hS[tid] = so * tanhf(cv);
        }
    }
    __syncthreads();
    if (tid < 2) {                        // dense head row for this batch
        float acc = dense_b[tid];
#pragma unroll
        for (int j = 0; j < Hn; ++j) acc += dense_w[tid * Hn + j] * hS[j];
        out[b * 2 + tid] = acc;
    }
}

extern "C" void kernel_launch(void* const* d_in, const int* in_sizes, int n_in,
                              void* d_out, int out_size, void* d_ws, size_t ws_size,
                              hipStream_t stream) {
    (void)in_sizes; (void)n_in; (void)out_size; (void)ws_size;
    const float* text    = (const float*)d_in[0];
    const float* Wx_w    = (const float*)d_in[1];
    const float* Wx_b    = (const float*)d_in[2];
    const float* Wxh_w   = (const float*)d_in[3];
    const float* Wxh_b   = (const float*)d_in[4];
    const float* att_w   = (const float*)d_in[5];
    const float* att_b   = (const float*)d_in[6];
    const float* W_ih    = (const float*)d_in[7];
    const float* W_hh    = (const float*)d_in[8];
    const float* b_ih    = (const float*)d_in[9];
    const float* b_hh    = (const float*)d_in[10];
    const float* dense_w = (const float*)d_in[11];
    const float* dense_b = (const float*)d_in[12];
    float* out = (float*)d_out;

    char* ws = (char*)d_ws;
    _Float16* Asw  = (_Float16*)(ws + OFF_ASW);
    _Float16* Bsw  = (_Float16*)(ws + OFF_BSW);
    float*    Wx   = (float*)   (ws + OFF_WX);
    float*    Wxh  = (float*)   (ws + OFF_WXHAT);
    float*    Xatt = (float*)   (ws + OFF_XATT);

    swizzle_b<<<(Bn * 16 * 2 * 32 * 16 + 255) / 256, 256, 0, stream>>>(text, Bsw);
    proj<<<(Bn * Ln * Nn + 255) / 256, 256, 0, stream>>>(text, Wx_w, Wx_b, Wxh_w, Wxh_b, Wx, Wxh);
    attn_scores<<<dim3(Ln / 16, Ln / 16, Bn), 256, 0, stream>>>(Wx, Wxh, att_w, att_b, Asw);
    xatt_wmma<<<dim3(Ln / 16, Bn), 64, 0, stream>>>(Asw, Bsw, Xatt);
    lstm_scan<<<Bn, 256, 0, stream>>>(Xatt, W_ih, W_hh, b_ih, b_hh, dense_w, dense_b, out);
}
